// StochasticDurationPredictor_14851996909637
// MI455X (gfx1250) — matrix operand order repaired and checked by reading, not compiled, not run on hardware
//
#include <hip/hip_runtime.h>
#include <hip/hip_bf16.h>
#include <math.h>

#define Bb 32
#define Cc 192
#define Tt 1024
#define NBIN 10
#define TAILC 5.0f
#define LOG2PI_F 1.8378770664093453f

typedef __attribute__((ext_vector_type(16))) _Float16 v16h;
typedef __attribute__((ext_vector_type(8)))  _Float16 v8h;
typedef __attribute__((ext_vector_type(8)))  float    v8f;

__device__ __forceinline__ unsigned packh2(float a, float b) {
  union { _Float16 h[2]; unsigned u; } p;
  p.h[0] = (_Float16)a; p.h[1] = (_Float16)b;
  return p.u;
}

// ---------------------------------------------------------------------------
// WMMA GEMM: Y[b, m, t] = sum_k W[m,k] * X[b,k,t] (+bias[m]) (*mask) (+add)
// K = 192 fixed. Block = 128 threads (4 waves). Block tile = 16(M) x 128(N);
// each wave computes two 16x16 N-subtiles sharing one A fragment.
// Double-buffered LDS, fully unrolled 6-step K loop (12 static WMMAs).
// grid = (T/128, ceil(M/16), B)
// ---------------------------------------------------------------------------
__global__ __launch_bounds__(128) void k_gemm(
    const float* __restrict__ W, const float* __restrict__ bias,
    const float* __restrict__ X, float* __restrict__ Y,
    const float* __restrict__ mask, const float* __restrict__ addsrc,
    int M, int outC, int useMask)
{
  __shared__ __align__(32) _Float16 sA0[16 * 32];
  __shared__ __align__(32) _Float16 sA1[16 * 32];
  __shared__ __align__(32) _Float16 sB0[128 * 32];
  __shared__ __align__(32) _Float16 sB1[128 * 32];

  const int tid  = threadIdx.x;
  const int b    = blockIdx.z;
  const int m0   = blockIdx.y * 16;
  const int t0   = blockIdx.x * 128;
  const int lane = tid & 31;
  const int wave = tid >> 5;
  const int hi   = lane >> 4;
  const int nl   = lane & 15;

  const int am = (tid * 4) >> 5;     // A stage: row
  const int ak = (tid * 4) & 31;     // A stage: k offset (multiple of 4)
  const int gm = m0 + am;

  const size_t xbase = ((size_t)b * Cc) * Tt + t0;

  v8f acc0 = {};
  v8f acc1 = {};

  float4 wv;        // A staging regs
  float  xr[32];    // B staging regs (one t-column, 32 k rows)

  // ---- prologue: global loads for k-step 0, store into buffer 0
  {
    if (gm < M) wv = *(const float4*)(W + (size_t)gm * Cc + ak);
    else        wv = make_float4(0.f, 0.f, 0.f, 0.f);
    const float* xp = X + xbase + tid;
#pragma unroll
    for (int c = 0; c < 32; ++c) xr[c] = xp[(size_t)c * Tt];

    uint2 au; au.x = packh2(wv.x, wv.y); au.y = packh2(wv.z, wv.w);
    *(uint2*)(sA0 + am * 32 + ak) = au;
#pragma unroll
    for (int j = 0; j < 16; ++j)
      *(unsigned*)(sB0 + tid * 32 + 2 * j) = packh2(xr[2 * j], xr[2 * j + 1]);
  }
  __syncthreads();

#pragma unroll
  for (int i = 0; i < 6; ++i) {
    const _Float16* cA = (i & 1) ? sA1 : sA0;
    const _Float16* cB = (i & 1) ? sB1 : sB0;
    _Float16* nA = (i & 1) ? sA0 : sA1;
    _Float16* nB = (i & 1) ? sB0 : sB1;

    // ---- issue global loads for next k-step (overlap with WMMAs below)
    if (i < 5) {
      int kk = (i + 1) * 32;
      if (gm < M) wv = *(const float4*)(W + (size_t)gm * Cc + kk + ak);
      else        wv = make_float4(0.f, 0.f, 0.f, 0.f);
      const float* xp = X + xbase + (size_t)kk * Tt + tid;
#pragma unroll
      for (int c = 0; c < 32; ++c) xr[c] = xp[(size_t)c * Tt];
    }

    // ---- fragments: contiguous k order, same in A and B
    // A 16x32 f16: M = lane&15, K = hi*16 + h
    v16h a = *(const v16h*)(cA + nl * 32 + hi * 16);
    // B 32x16 f16: N = lane&15 (col t), K = hi*16 + h
    v16h b0 = *(const v16h*)(cB + (wave * 32 + nl) * 32 + hi * 16);
    v16h b1 = *(const v16h*)(cB + (wave * 32 + 16 + nl) * 32 + hi * 16);

    acc0 = __builtin_amdgcn_wmma_f32_16x16x32_f16(
        false, a, false, b0, (short)0, acc0, false, false);
    acc1 = __builtin_amdgcn_wmma_f32_16x16x32_f16(
        false, a, false, b1, (short)0, acc1, false, false);

    // ---- convert + store next tile into the other buffer
    if (i < 5) {
      uint2 au; au.x = packh2(wv.x, wv.y); au.y = packh2(wv.z, wv.w);
      *(uint2*)(nA + am * 32 + ak) = au;
#pragma unroll
      for (int j = 0; j < 16; ++j)
        *(unsigned*)(nB + tid * 32 + 2 * j) = packh2(xr[2 * j], xr[2 * j + 1]);
    }
    __syncthreads();
  }

  // ---- epilogue: D layout = vgpr r -> row r + 8*hi, col = nl
  const int ta = t0 + wave * 32 + nl;
  const int tb = ta + 16;
  float mva = 1.0f, mvb = 1.0f;
  if (useMask) {
    mva = mask[(size_t)b * Tt + ta];
    mvb = mask[(size_t)b * Tt + tb];
  }
#pragma unroll
  for (int r = 0; r < 8; ++r) {
    int gmo = m0 + hi * 8 + r;
    if (gmo < M) {
      float v0 = acc0[r] + bias[gmo];
      float v1 = acc1[r] + bias[gmo];
      if (useMask) { v0 *= mva; v1 *= mvb; }
      if (addsrc) {
        v0 += addsrc[((size_t)b * Cc + gmo) * Tt + ta];
        v1 += addsrc[((size_t)b * Cc + gmo) * Tt + tb];
      }
      Y[((size_t)b * outC + gmo) * Tt + ta] = v0;
      Y[((size_t)b * outC + gmo) * Tt + tb] = v1;
    }
  }
}

// ---------------------------------------------------------------------------
// Part 1 of a DDS layer: y = gelu(LN_c(depthwise(x*mask, dil)))
// grid = (T/64, 1, B), block = 256 (4 channel-groups x 64 t)
// ---------------------------------------------------------------------------
__global__ __launch_bounds__(256) void k_dw_ln_gelu(
    const float* __restrict__ X, const float* __restrict__ mask,
    const float* __restrict__ kw, const float* __restrict__ kb,
    const float* __restrict__ gamma, const float* __restrict__ beta,
    float* __restrict__ Y, int dil)
{
  __shared__ float redS[4][64];
  __shared__ float redQ[4][64];
  __shared__ float smean[64];
  __shared__ float srstd[64];
  const int tid = threadIdx.x;
  const int b   = blockIdx.z;
  const int t0  = blockIdx.x * 64;
  const int tt2 = tid & 63;
  const int cq  = tid >> 6;
  const int t   = t0 + tt2;
  const float* mrow = mask + (size_t)b * Tt;

  float vals[48];
  float s = 0.f, q = 0.f;
#pragma unroll
  for (int i = 0; i < 48; ++i) {
    int c = cq * 48 + i;
    const float* xrow = X + ((size_t)b * Cc + c) * Tt;
    float a = kb[c];
#pragma unroll
    for (int j = 0; j < 3; ++j) {
      int ts = t + (j - 1) * dil;
      if (ts >= 0 && ts < Tt) a += kw[c * 3 + j] * xrow[ts] * mrow[ts];
    }
    vals[i] = a; s += a; q += a * a;
  }
  redS[cq][tt2] = s; redQ[cq][tt2] = q;
  __syncthreads();
  if (tid < 64) {
    float ts = 0.f, tq = 0.f;
#pragma unroll
    for (int k = 0; k < 4; ++k) { ts += redS[k][tid]; tq += redQ[k][tid]; }
    float mean = ts / (float)Cc;
    float var  = tq / (float)Cc - mean * mean;
    smean[tid] = mean;
    srstd[tid] = rsqrtf(var + 1e-5f);
  }
  __syncthreads();
  float mean = smean[tt2], rstd = srstd[tt2];
#pragma unroll
  for (int i = 0; i < 48; ++i) {
    int c = cq * 48 + i;
    float v = (vals[i] - mean) * rstd * gamma[c] + beta[c];
    float g = 0.5f * v * (1.0f + erff(v * 0.70710678f));
    Y[((size_t)b * Cc + c) * Tt + t] = g;
  }
}

// ---------------------------------------------------------------------------
// Part 2 of a DDS layer: cur += gelu(LN_c(X))
// ---------------------------------------------------------------------------
__global__ __launch_bounds__(256) void k_ln_gelu_add(
    const float* __restrict__ X, const float* __restrict__ gamma,
    const float* __restrict__ beta, float* __restrict__ cur)
{
  __shared__ float redS[4][64];
  __shared__ float redQ[4][64];
  __shared__ float smean[64];
  __shared__ float srstd[64];
  const int tid = threadIdx.x;
  const int b   = blockIdx.z;
  const int t0  = blockIdx.x * 64;
  const int tt2 = tid & 63;
  const int cq  = tid >> 6;
  const int t   = t0 + tt2;

  float vals[48];
  float s = 0.f, q = 0.f;
#pragma unroll
  for (int i = 0; i < 48; ++i) {
    int c = cq * 48 + i;
    float a = X[((size_t)b * Cc + c) * Tt + t];
    vals[i] = a; s += a; q += a * a;
  }
  redS[cq][tt2] = s; redQ[cq][tt2] = q;
  __syncthreads();
  if (tid < 64) {
    float ts = 0.f, tq = 0.f;
#pragma unroll
    for (int k = 0; k < 4; ++k) { ts += redS[k][tid]; tq += redQ[k][tid]; }
    float mean = ts / (float)Cc;
    float var  = tq / (float)Cc - mean * mean;
    smean[tid] = mean;
    srstd[tid] = rsqrtf(var + 1e-5f);
  }
  __syncthreads();
  float mean = smean[tt2], rstd = srstd[tt2];
#pragma unroll
  for (int i = 0; i < 48; ++i) {
    int c = cq * 48 + i;
    float v = (vals[i] - mean) * rstd * gamma[c] + beta[c];
    float g = 0.5f * v * (1.0f + erff(v * 0.70710678f));
    cur[((size_t)b * Cc + c) * Tt + t] += g;
  }
}

// ---------------------------------------------------------------------------
// out[b,c,t] = wv[c] * src[b, t (srcStride)] + bv[c] (+ g[b,c,t])
// grid = (T/256, C, B)
// ---------------------------------------------------------------------------
__global__ void k_bcast(const float* __restrict__ wv, const float* __restrict__ bv,
                        const float* __restrict__ src, int srcStride,
                        const float* __restrict__ g, float* __restrict__ out)
{
  int t = blockIdx.x * 256 + threadIdx.x;
  int c = blockIdx.y; int b = blockIdx.z;
  float sv = src[(size_t)b * srcStride + t];
  float v = wv[c] * sv + bv[c];
  size_t idx = ((size_t)b * Cc + c) * Tt + t;
  if (g) v += g[idx];
  out[idx] = v;
}

// ---------------------------------------------------------------------------
// z_q init (post elementwise affine) + logq base + ldq init.  grid = B x 256
// ---------------------------------------------------------------------------
__global__ __launch_bounds__(256) void k_ea_init(
    const float* __restrict__ eq, const float* __restrict__ mask,
    const float* __restrict__ m2, const float* __restrict__ logs2,
    float* __restrict__ zq, float* __restrict__ accLogq,
    float* __restrict__ accLdq, float* __restrict__ accLogdet)
{
  __shared__ float r1[256], r2[256];
  int tid = threadIdx.x; int b = blockIdx.x;
  float e0 = expf(logs2[0]), e1 = expf(logs2[1]);
  float lq = 0.f, ms = 0.f;
  for (int t = tid; t < Tt; t += 256) {
    float mk = mask[(size_t)b * Tt + t];
    float q0 = eq[((size_t)b * 2 + 0) * Tt + t] * mk;
    float q1 = eq[((size_t)b * 2 + 1) * Tt + t] * mk;
    zq[((size_t)b * 2 + 0) * Tt + t] = (m2[0] + e0 * q0) * mk;
    zq[((size_t)b * 2 + 1) * Tt + t] = (m2[1] + e1 * q1) * mk;
    lq += -0.5f * (LOG2PI_F + q0 * q0) * mk - 0.5f * (LOG2PI_F + q1 * q1) * mk;
    ms += mk;
  }
  r1[tid] = lq; r2[tid] = ms; __syncthreads();
  for (int s2 = 128; s2 > 0; s2 >>= 1) {
    if (tid < s2) { r1[tid] += r1[tid + s2]; r2[tid] += r2[tid + s2]; }
    __syncthreads();
  }
  if (tid == 0) {
    accLogq[b]   = r1[0];
    accLdq[b]    = (logs2[0] + logs2[1]) * r2[0];
    accLogdet[b] = 0.f;
  }
}

__device__ __forceinline__ float softplusf(float x) {
  return x > 0.f ? x + log1pf(expf(-x)) : log1pf(expf(x));
}
__device__ __forceinline__ float logsigf(float x) {
  return x >= 0.f ? -log1pf(expf(-x)) : x - log1pf(expf(x));
}
__device__ __forceinline__ void softmax_widths(const float (&u)[NBIN],
                                               float (&cum)[NBIN + 1],
                                               float (&wid)[NBIN]) {
  float mx = u[0];
#pragma unroll
  for (int j = 1; j < NBIN; ++j) mx = fmaxf(mx, u[j]);
  float e[NBIN]; float sum = 0.f;
#pragma unroll
  for (int j = 0; j < NBIN; ++j) { e[j] = expf(u[j] - mx); sum += e[j]; }
  float inv = 1.f / sum;
  cum[0] = -TAILC;
  float run = 0.f;
#pragma unroll
  for (int j = 0; j < NBIN; ++j) {
    run += 0.001f + 0.99f * e[j] * inv;          // MIN_W + (1-MIN_W*NB)*p
    cum[j + 1] = run * (2.0f * TAILC) - TAILC;
  }
  cum[NBIN] = TAILC;
#pragma unroll
  for (int j = 0; j < NBIN; ++j) wid[j] = cum[j + 1] - cum[j];
}

// ---------------------------------------------------------------------------
// RQ spline flow step (operates in-place on 2-ch z, writes flipped output),
// accumulates sum(lad*mask) into acc[b].  grid = B x 1024
// ---------------------------------------------------------------------------
__global__ __launch_bounds__(1024) void k_spline(
    const float* __restrict__ H, const float* __restrict__ mask,
    float* __restrict__ z, float* __restrict__ accLad)
{
  __shared__ float red[1024];
  int t = threadIdx.x; int b = blockIdx.x;
  float mk = mask[(size_t)b * Tt + t];
  float x0 = z[((size_t)b * 2 + 0) * Tt + t];
  float x1 = z[((size_t)b * 2 + 1) * Tt + t];
  const float s = 0.07216878f;  // 1/sqrt(192)

  float hw[NBIN], hh[NBIN], hd[NBIN - 1];
#pragma unroll
  for (int j = 0; j < NBIN; ++j) hw[j] = H[((size_t)b * 32 + j) * Tt + t] * s;
#pragma unroll
  for (int j = 0; j < NBIN; ++j) hh[j] = H[((size_t)b * 32 + 10 + j) * Tt + t] * s;
#pragma unroll
  for (int j = 0; j < NBIN - 1; ++j) hd[j] = H[((size_t)b * 32 + 20 + j) * Tt + t];

  float cw[NBIN + 1], ww[NBIN], ch[NBIN + 1], hn[NBIN], dd[NBIN + 1];
  softmax_widths(hw, cw, ww);
  softmax_widths(hh, ch, hn);
  dd[0] = 1.0f; dd[NBIN] = 1.0f;   // MIN_D + softplus(const) == 1 exactly
#pragma unroll
  for (int j = 0; j < NBIN - 1; ++j) dd[1 + j] = 0.001f + softplusf(hd[j]);

  float x = fminf(fmaxf(x1, -TAILC), TAILC);
  int cnt = 0;
#pragma unroll
  for (int k = 0; k < NBIN; ++k) cnt += (x >= cw[k]) ? 1 : 0;
  // locs[NBIN] = TAIL + 1e-6 can never be <= x, so it never contributes.
  int idx = cnt - 1;
  idx = idx < 0 ? 0 : (idx > NBIN - 1 ? NBIN - 1 : idx);

  float icw = 0.f, ibw = 1.f, ich = 0.f, ibh = 1.f, d0 = 1.f, d1 = 1.f;
#pragma unroll
  for (int k = 0; k < NBIN; ++k) {
    if (k == idx) {
      icw = cw[k]; ibw = ww[k]; ich = ch[k]; ibh = hn[k];
      d0 = dd[k]; d1 = dd[k + 1];
    }
  }
  float idelta = ibh / ibw;
  float th  = (x - icw) / ibw;
  float tt2 = th * (1.f - th);
  float numer = ibh * (idelta * th * th + d0 * tt2);
  float denom = idelta + (d0 + d1 - 2.f * idelta) * tt2;
  float outv = ich + numer / denom;
  float dnum = idelta * idelta *
               (d1 * th * th + 2.f * idelta * tt2 + d0 * (1.f - th) * (1.f - th));
  float lad = logf(dnum) - 2.f * logf(denom);
  bool inside = (x1 >= -TAILC) && (x1 <= TAILC);
  float x1n = inside ? outv : x1;
  lad = inside ? lad : 0.f;

  // write flipped (z = concat(x0, x1n)*mask, then [:, ::-1])
  z[((size_t)b * 2 + 0) * Tt + t] = x1n * mk;
  z[((size_t)b * 2 + 1) * Tt + t] = x0 * mk;

  red[t] = lad * mk; __syncthreads();
  for (int s2 = 512; s2 > 0; s2 >>= 1) {
    if (t < s2) red[t] += red[t + s2];
    __syncthreads();
  }
  if (t == 0) accLad[b] += red[0];
}

// ---------------------------------------------------------------------------
// split u, log transform, elementwise affine.  grid = B x 1024
// ---------------------------------------------------------------------------
__global__ __launch_bounds__(1024) void k_split_ea(
    const float* __restrict__ zq, const float* __restrict__ wdur,
    const float* __restrict__ mask, const float* __restrict__ m2,
    const float* __restrict__ logs2, float* __restrict__ zb,
    float* __restrict__ accLdq, float* __restrict__ accLogdet)
{
  __shared__ float r1[1024], r2[1024];
  int t = threadIdx.x; int b = blockIdx.x;
  float mk = mask[(size_t)b * Tt + t];
  float zu = zq[((size_t)b * 2 + 0) * Tt + t];
  float z1 = zq[((size_t)b * 2 + 1) * Tt + t];
  float u  = (1.f / (1.f + expf(-zu))) * mk;
  float z0 = (wdur[(size_t)b * Tt + t] - u) * mk;
  float ldqc = (logsigf(zu) + logsigf(-zu)) * mk;
  float z0l = logf(fmaxf(z0, 1e-5f)) * mk;
  float ldetc = -z0l;
  float za  = (m2[0] + expf(logs2[0]) * z0l) * mk;
  float zbv = (m2[1] + expf(logs2[1]) * z1) * mk;
  ldetc += (logs2[0] + logs2[1]) * mk;
  zb[((size_t)b * 2 + 0) * Tt + t] = za;
  zb[((size_t)b * 2 + 1) * Tt + t] = zbv;
  r1[t] = ldqc; r2[t] = ldetc; __syncthreads();
  for (int s2 = 512; s2 > 0; s2 >>= 1) {
    if (t < s2) { r1[t] += r1[t + s2]; r2[t] += r2[t + s2]; }
    __syncthreads();
  }
  if (t == 0) { accLdq[b] += r1[0]; accLogdet[b] += r2[0]; }
}

// ---------------------------------------------------------------------------
// final nll + logq.  grid = B x 1024
// ---------------------------------------------------------------------------
__global__ __launch_bounds__(1024) void k_final(
    const float* __restrict__ zb, const float* __restrict__ mask,
    const float* __restrict__ accLogq, const float* __restrict__ accLdq,
    const float* __restrict__ accLogdet, float* __restrict__ out)
{
  __shared__ float red[1024];
  int t = threadIdx.x; int b = blockIdx.x;
  float mk = mask[(size_t)b * Tt + t];
  float z0 = zb[((size_t)b * 2 + 0) * Tt + t];
  float z1 = zb[((size_t)b * 2 + 1) * Tt + t];
  red[t] = 0.5f * (LOG2PI_F + z0 * z0) * mk + 0.5f * (LOG2PI_F + z1 * z1) * mk;
  __syncthreads();
  for (int s2 = 512; s2 > 0; s2 >>= 1) {
    if (t < s2) red[t] += red[t + s2];
    __syncthreads();
  }
  if (t == 0) out[b] = red[0] - accLogdet[b] + accLogq[b] - accLdq[b];
}

// ---------------------------------------------------------------------------
// Host side
// ---------------------------------------------------------------------------
extern "C" void kernel_launch(void* const* d_in, const int* in_sizes, int n_in,
                              void* d_out, int out_size, void* d_ws, size_t ws_size,
                              hipStream_t stream)
{
  (void)in_sizes; (void)n_in; (void)out_size; (void)ws_size;
  auto P = [&](int i) -> const float* { return (const float*)d_in[i]; };
  const float* x    = P(0);
  const float* mask = P(1);
  const float* wdur = P(2);
  const float* eq   = P(3);

  const size_t SZ = (size_t)Bb * Cc * Tt;
  float* ws    = (float*)d_ws;
  float* cur   = ws;
  float* t1    = ws + SZ;
  float* t2    = ws + 2 * SZ;
  float* xproj = ws + 3 * SZ;
  float* gpost = ws + 4 * SZ;
  float* h29   = ws + 5 * SZ;                       // B x 32 x T
  float* zq    = h29 + (size_t)Bb * 32 * Tt;        // B x 2 x T
  float* zb    = zq + (size_t)Bb * 2 * Tt;          // B x 2 x T
  float* accLogq   = zb + (size_t)Bb * 2 * Tt;
  float* accLdq    = accLogq + Bb;
  float* accLogdet = accLdq + Bb;

  dim3 gG(Tt / 128, Cc / 16, Bb), bG(128);
  dim3 gG29(Tt / 128, 2, Bb);
  dim3 gP(Tt / 64, 1, Bb), bP(256);
  dim3 gBc(Tt / 256, Cc, Bb), bBc(256);

  auto dds3 = [&](int baseIdx) {
    int dil = 1;
    for (int l = 0; l < 3; ++l) {
      int bi = baseIdx + 8 * l;
      k_dw_ln_gelu<<<gP, bP, 0, stream>>>(cur, mask, P(bi + 0), P(bi + 1),
                                          P(bi + 2), P(bi + 3), t1, dil);
      k_gemm<<<gG, bG, 0, stream>>>(P(bi + 4), P(bi + 5), t1, t2,
                                    nullptr, nullptr, Cc, Cc, 0);
      k_ln_gelu_add<<<gP, bP, 0, stream>>>(t2, P(bi + 6), P(bi + 7), cur);
      dil *= 3;
    }
  };

  // Phase A: main x path
  k_gemm<<<gG, bG, 0, stream>>>(P(4), P(5), x, cur, nullptr, nullptr, Cc, Cc, 0);
  dds3(6);
  k_gemm<<<gG, bG, 0, stream>>>(P(30), P(31), cur, xproj, mask, nullptr, Cc, Cc, 1);

  // Phase B: h_w path, g_post = xproj + h_w (fused in epilogue)
  k_bcast<<<gBc, bBc, 0, stream>>>(P(32), P(33), wdur, Tt, nullptr, cur);
  dds3(34);
  k_gemm<<<gG, bG, 0, stream>>>(P(58), P(59), cur, gpost, mask, xproj, Cc, Cc, 1);

  // Phase C: z_q init
  k_ea_init<<<Bb, 256, 0, stream>>>(eq, mask, P(62), P(63), zq,
                                    accLogq, accLdq, accLogdet);

  // Phase D: 4 post_cf flows (params at 176 + 28*f)
  for (int f = 0; f < 4; ++f) {
    int base = 176 + 28 * f;
    k_bcast<<<gBc, bBc, 0, stream>>>(P(base + 0), P(base + 1), zq, 2 * Tt,
                                     gpost, cur);
    dds3(base + 2);
    k_gemm<<<gG29, bG, 0, stream>>>(P(base + 26), P(base + 27), cur, h29,
                                    mask, nullptr, 29, 32, 1);
    k_spline<<<Bb, 1024, 0, stream>>>(h29, mask, zq, accLdq);
  }

  // Phase E: split u / log / elementwise affine
  k_split_ea<<<Bb, 1024, 0, stream>>>(zq, wdur, mask, P(60), P(61), zb,
                                      accLdq, accLogdet);

  // Phase F: 4 cf flows (params at 64 + 28*f), g = xproj
  for (int f = 0; f < 4; ++f) {
    int base = 64 + 28 * f;
    k_bcast<<<gBc, bBc, 0, stream>>>(P(base + 0), P(base + 1), zb, 2 * Tt,
                                     xproj, cur);
    dds3(base + 2);
    k_gemm<<<gG29, bG, 0, stream>>>(P(base + 26), P(base + 27), cur, h29,
                                    mask, nullptr, 29, 32, 1);
    k_spline<<<Bb, 1024, 0, stream>>>(h29, mask, zb, accLogdet);
  }

  // Phase G
  k_final<<<Bb, 1024, 0, stream>>>(zb, mask, accLogq, accLdq, accLogdet,
                                   (float*)d_out);
}
